// DynamicsPredictor_46377056862522
// MI455X (gfx1250) — matrix operand order repaired and checked by reading, not compiled, not run on hardware
//
#include <hip/hip_runtime.h>

// ---------------- problem constants (match reference) ----------------
#define BB      2
#define NP_     2048
#define FLOOR_  64
#define NTOOL_  512
#define NN_     2624          // NP_ + FLOOR_ + NTOOL_
#define NREL_   12288
#define DS_     3
#define NF_     128
#define NFH_    256
#define PIN_    275           // 3 + 1 + 15 + 256
#define KPA_    288           // PIN_ padded to 32
#define RIN_    566           // 2*275 + 15 + 1
#define KPR_    576           // RIN_ padded to 32
#define BN_     (BB*NN_)      // 5248  (multiple of 128)
#define BR_     (BB*NREL_)    // 24576 (multiple of 128)
#define TOUCH_THRESH (NP_ + FLOOR_)   // 2112
#define BOUND_  0.005f

typedef __attribute__((ext_vector_type(16))) _Float16 v16h;
typedef __attribute__((ext_vector_type(8)))  _Float16 v8h;
typedef __attribute__((ext_vector_type(8)))  float    v8f;

// ---------------- K1: decode one-hot rows -> indices (+touch) --------
__global__ __launch_bounds__(32) void decode_onehot(
    const float* __restrict__ R, int* __restrict__ idx,
    float* __restrict__ touch) {
  const int row  = blockIdx.x;            // 0..BR_-1
  const int lane = threadIdx.x;           // 0..31
  const float* p = R + (long)row * NN_;
  int found = 0;
  for (int c = lane; c < NN_; c += 32)
    if (p[c] > 0.5f) found = c;
  #pragma unroll
  for (int off = 16; off > 0; off >>= 1)
    found = max(found, __shfl_down(found, off, 32));
  if (lane == 0) {
    idx[row] = found;
    if (touch != nullptr && found >= TOUCH_THRESH)
      touch[row / NREL_] = 1.0f;          // benign race: write 1.0
  }
}

// ---------------- K2: state_norm_t  (B,N,15) -------------------------
__global__ __launch_bounds__(256) void build_state_norm(
    const float* __restrict__ state_cur, const float* __restrict__ action_his,
    const float* __restrict__ mean_p, const float* __restrict__ std_p,
    const float* __restrict__ mean_d, const float* __restrict__ std_d,
    float* __restrict__ snorm) {
  int i = blockIdx.x * blockDim.x + threadIdx.x;     // over BN_
  if (i >= BN_) return;
  int b = i / NN_, n = i - b * NN_;
  float* out = snorm + (long)i * 15;
  for (int t = 0; t < 4; ++t)
    for (int d = 0; d < 3; ++d) {
      float v = 0.f;
      if (n >= TOUCH_THRESH) {
        int tool = n - TOUCH_THRESH;
        long i1 = (((long)b * 5 + (t + 1)) * NTOOL_ + tool) * 3 + d;
        long i0 = (((long)b * 5 + t) * NTOOL_ + tool) * 3 + d;
        v = (action_his[i1] - action_his[i0] - mean_d[d]) / std_d[d];
      }
      out[t * 3 + d] = v;
    }
  for (int d = 0; d < 3; ++d)
    out[12 + d] = (state_cur[(long)i * 3 + d] - mean_p[d]) / std_p[d];
}

// ---------------- K3: instance-center reduction ----------------------
__global__ __launch_bounds__(256) void center_accum(
    const float* __restrict__ snorm, const float* __restrict__ p_inst,
    float* __restrict__ centerSum) {
  int b = blockIdx.x, tid = threadIdx.x;
  float loc[16];
  for (int d = 0; d < 16; ++d) loc[d] = 0.f;
  for (int p = tid; p < NP_; p += blockDim.x) {
    float w = p_inst[(long)b * NP_ + p];
    const float* s = snorm + ((long)b * NN_ + p) * 15;
    for (int d = 0; d < 15; ++d) loc[d] += w * s[d];
    loc[15] += w;
  }
  for (int d = 0; d < 16; ++d) atomicAdd(&centerSum[b * 16 + d], loc[d]);
}

__global__ __launch_bounds__(32) void center_fin(
    const float* __restrict__ centerSum, float* __restrict__ centerFin) {
  int i = threadIdx.x;
  if (i < BB * 16) {
    int b = i >> 4, d = i & 15;
    centerFin[i] = (d < 15) ? centerSum[i] / (centerSum[b * 16 + 15] + 1e-6f)
                            : 0.f;
  }
}

// ---------------- K4: particle feature matrix A (f16, K padded) ------
__global__ __launch_bounds__(288) void build_A(
    const float* __restrict__ attrs, const float* __restrict__ physics,
    const float* __restrict__ p_inst, const float* __restrict__ p_rigid,
    const float* __restrict__ snorm, const float* __restrict__ centerFin,
    const float* __restrict__ memory, _Float16* __restrict__ A16,
    float* __restrict__ gvals) {
  int row = blockIdx.x;                 // b*NN_+n
  int b = row / NN_, n = row - b * NN_;
  int t = threadIdx.x;                  // 0..287
  float v = 0.f;
  if (t < 3) v = attrs[(long)row * 3 + t];
  else if (t == 3) v = (n < NP_) ? physics[(long)b * NP_ + n] : 0.f;
  else if (t < 19) {
    int d = t - 4;
    if (n < NP_) {
      float pi   = p_inst[(long)b * NP_ + n];
      float prpp = pi * p_rigid[b];
      float sp   = snorm[(long)row * 15 + d];
      float cpp  = pi * centerFin[b * 16 + d];
      float c    = (1.f - prpp) * sp + prpp * cpp;
      v = sp - c;
    }
  } else if (t < PIN_) {
    int m = t - 19, layer = m >> 7, f = m & 127;
    v = memory[(((long)b * 2 + layer) * NN_ + n) * NF_ + f];
  }
  A16[(long)row * KPA_ + t] = (_Float16)v;
  if (t == 0) gvals[row] = (n < NP_) ? p_inst[(long)b * NP_ + n] : 0.f;
}

// ---------------- K5: weight f32 -> f16 with K padding ----------------
__global__ __launch_bounds__(256) void cvt_pad(
    const float* __restrict__ src, _Float16* __restrict__ dst,
    int rows, int K, int Kp) {
  int i = blockIdx.x * blockDim.x + threadIdx.x;
  if (i >= rows * Kp) return;
  int r = i / Kp, c = i - r * Kp;
  dst[i] = (_Float16)((c < K) ? src[(long)r * K + c] : 0.f);
}

// ---------------- K6: relation feature matrix (f16, K padded) --------
__global__ __launch_bounds__(192) void build_rin(
    const _Float16* __restrict__ A16, const float* __restrict__ snorm,
    const float* __restrict__ gvals, const int* __restrict__ ridx,
    const int* __restrict__ sidx, _Float16* __restrict__ Rin) {
  int r = blockIdx.x;
  int b = r / NREL_;
  long nr = (long)b * NN_ + ridx[r];
  long ns = (long)b * NN_ + sidx[r];
  for (int t = threadIdx.x; t < KPR_; t += blockDim.x) {
    _Float16 v = (_Float16)0.f;
    if (t < PIN_)            v = A16[nr * KPA_ + t];
    else if (t < 2 * PIN_)   v = A16[ns * KPA_ + (t - PIN_)];
    else if (t < 2 * PIN_ + 15) {
      int d = t - 2 * PIN_;
      v = (_Float16)(snorm[nr * 15 + d] - snorm[ns * 15 + d]);
    } else if (t == 2 * PIN_ + 15) {
      v = (_Float16)fabsf(gvals[nr] - gvals[ns]);
    }
    Rin[(long)r * KPR_ + t] = v;
  }
}

// ---------------- K7: WMMA GEMM  Y = relu(X @ W^T + b)  (f16 in/out) --
// X:[M,Kp] row-major f16, W:[Nout,Kp] row-major f16.
// Register-blocked 2x2: each wave computes a 32x32 output tile with 4
// accumulators, reusing each A/B fragment twice (halves L2 traffic per WMMA).
__device__ __forceinline__ v16h load_fragA(const _Float16* p) {
  v8h lo = *(const v8h*)(p);            // K = kA .. kA+7
  v8h hi = *(const v8h*)(p + 16);       // K = kA+16 .. kA+23
  return __builtin_shufflevector(lo, hi, 0,1,2,3,4,5,6,7,8,9,10,11,12,13,14,15);
}
__device__ __forceinline__ v16h load_fragB(const _Float16* p) {
  v8h lo = *(const v8h*)(p);            // K = kB .. kB+7
  v8h hi = *(const v8h*)(p + 16);       // K = kB+16 .. kB+23
  return __builtin_shufflevector(lo, hi, 0,1,2,3,4,5,6,7,8,9,10,11,12,13,14,15);
}

__global__ __launch_bounds__(128) void gemm_f16_wmma(
    const _Float16* __restrict__ X, const _Float16* __restrict__ W,
    const float* __restrict__ bias, _Float16* __restrict__ Y,
    int M, int Kp, int Nout, int relu) {
  const int lane = threadIdx.x & 31;
  const int wave = threadIdx.x >> 5;
  const int row0 = (blockIdx.x * 4 + wave) << 5;   // 32 rows per wave
  const int col0 = blockIdx.y << 5;                // 32 cols per wave
  if (row0 >= M) return;                           // wave-uniform

  const int l16 = lane & 15;
  const int hi  = lane >> 4;                       // 0 or 1
  const int kA  = hi << 3;                         // A half: 0 / 8
  const int kB  = hi << 3;                         // B half: 0 / 8
  const _Float16* xp0 = X + (long)(row0 + l16) * Kp + kA;
  const _Float16* xp1 = xp0 + (long)16 * Kp;
  const _Float16* wp0 = W + (long)(col0 + l16) * Kp + kB;
  const _Float16* wp1 = wp0 + (long)16 * Kp;

  v8f acc00 = {0.f,0.f,0.f,0.f,0.f,0.f,0.f,0.f};
  v8f acc01 = {0.f,0.f,0.f,0.f,0.f,0.f,0.f,0.f};
  v8f acc10 = {0.f,0.f,0.f,0.f,0.f,0.f,0.f,0.f};
  v8f acc11 = {0.f,0.f,0.f,0.f,0.f,0.f,0.f,0.f};
  for (int k0 = 0; k0 < Kp; k0 += 32) {
    v16h a0 = load_fragA(xp0 + k0);
    v16h a1 = load_fragA(xp1 + k0);
    v16h b0 = load_fragB(wp0 + k0);
    v16h b1 = load_fragB(wp1 + k0);
    acc00 = __builtin_amdgcn_wmma_f32_16x16x32_f16(false, a0, false, b0, (short)0, acc00, false, false);
    acc01 = __builtin_amdgcn_wmma_f32_16x16x32_f16(false, a0, false, b1, (short)0, acc01, false, false);
    acc10 = __builtin_amdgcn_wmma_f32_16x16x32_f16(false, a1, false, b0, (short)0, acc10, false, false);
    acc11 = __builtin_amdgcn_wmma_f32_16x16x32_f16(false, a1, false, b1, (short)0, acc11, false, false);
  }
  // epilogue: D layout -> lane l, VGPR v holds (M = v + 8*(l>=16), N = l%16)
  const float bv0 = bias ? bias[col0 + l16] : 0.f;
  const float bv1 = bias ? bias[col0 + 16 + l16] : 0.f;
  const int rb = (hi << 3);
  #pragma unroll
  for (int v = 0; v < 8; ++v) {
    float v00 = acc00[v] + bv0, v01 = acc01[v] + bv1;
    float v10 = acc10[v] + bv0, v11 = acc11[v] + bv1;
    if (relu) {
      v00 = fmaxf(v00, 0.f); v01 = fmaxf(v01, 0.f);
      v10 = fmaxf(v10, 0.f); v11 = fmaxf(v11, 0.f);
    }
    long r0 = (long)(row0 + rb + v) * Nout;
    long r1 = (long)(row0 + 16 + rb + v) * Nout;
    Y[r0 + col0 + l16]      = (_Float16)v00;
    Y[r0 + col0 + 16 + l16] = (_Float16)v01;
    Y[r1 + col0 + l16]      = (_Float16)v10;
    Y[r1 + col0 + 16 + l16] = (_Float16)v11;
  }
}

// ---------------- K8: propagation-step helpers -----------------------
__global__ __launch_bounds__(128) void build_erin(
    const _Float16* __restrict__ Renc, const _Float16* __restrict__ Eff,
    const int* __restrict__ ridx, const int* __restrict__ sidx,
    _Float16* __restrict__ Erin) {
  int r = blockIdx.x;
  int b = r / NREL_;
  long nr = (long)b * NN_ + ridx[r];
  long ns = (long)b * NN_ + sidx[r];
  for (int t = threadIdx.x; t < 384; t += blockDim.x) {
    _Float16 v;
    if (t < 128)      v = Renc[(long)r * NF_ + t];
    else if (t < 256) v = Eff[nr * NF_ + (t - 128)];
    else              v = Eff[ns * NF_ + (t - 256)];
    Erin[(long)r * 384 + t] = v;
  }
}

__global__ __launch_bounds__(256) void scatter_add(
    const _Float16* __restrict__ Erel, const int* __restrict__ ridx,
    float* __restrict__ Agg) {
  int i = blockIdx.x * blockDim.x + threadIdx.x;   // BR_*128
  if (i >= BR_ * NF_) return;
  int r = i >> 7, c = i & 127;
  int b = r / NREL_;
  atomicAdd(&Agg[((long)b * NN_ + ridx[r]) * NF_ + c], (float)Erel[i]);
}

__global__ __launch_bounds__(256) void build_p2in(
    const _Float16* __restrict__ Penc, const float* __restrict__ Agg,
    _Float16* __restrict__ P2in) {
  int i = blockIdx.x * blockDim.x + threadIdx.x;   // BN_*256
  if (i >= BN_ * 256) return;
  int row = i >> 8, c = i & 255;
  P2in[i] = (c < 128) ? Penc[(long)row * NF_ + c]
                      : (_Float16)Agg[(long)row * NF_ + (c - 128)];
}

// ---------------- K9: rigid branch ------------------------------------
__global__ __launch_bounds__(128) void inst_eff_reduce(
    const _Float16* __restrict__ Eff, const float* __restrict__ p_inst,
    float* __restrict__ instEff) {
  int b = blockIdx.x, c = threadIdx.x;
  float s = 0.f;
  for (int p = 0; p < NP_; ++p)
    s += p_inst[(long)b * NP_ + p] * (float)Eff[((long)b * NN_ + p) * NF_ + c];
  instEff[b * NF_ + c] = s;
}

__global__ __launch_bounds__(128) void rigid_mlp(
    const float* __restrict__ instEff,
    const float* __restrict__ w0, const float* __restrict__ b0,
    const float* __restrict__ w1, const float* __restrict__ b1,
    const float* __restrict__ w2, const float* __restrict__ b2,
    const float* __restrict__ centerFin,
    const float* __restrict__ mean_p, const float* __restrict__ std_p,
    const float* __restrict__ mean_d, const float* __restrict__ std_d,
    float* __restrict__ xform) {
  __shared__ float x[128], h[128], o7[7];
  int b = blockIdx.x, t = threadIdx.x;
  x[t] = instEff[b * 128 + t];
  __syncthreads();
  float s = b0[t];
  for (int k = 0; k < 128; ++k) s += w0[t * 128 + k] * x[k];
  h[t] = fmaxf(s, 0.f);
  __syncthreads();
  s = b1[t];
  for (int k = 0; k < 128; ++k) s += w1[t * 128 + k] * h[k];
  x[t] = fmaxf(s, 0.f);
  __syncthreads();
  if (t < 7) {
    s = b2[t];
    for (int k = 0; k < 128; ++k) s += w2[t * 128 + k] * x[k];
    o7[t] = s;
  }
  __syncthreads();
  if (t == 0) {
    float r = o7[0] + 1.f, i = o7[1], j = o7[2], k = o7[3];
    float two_s = 2.f / (r * r + i * i + j * j + k * k);
    float* X = xform + b * 16;
    X[0] = 1.f - two_s * (j * j + k * k); X[1] = two_s * (i * j - k * r); X[2] = two_s * (i * k + j * r);
    X[3] = two_s * (i * j + k * r); X[4] = 1.f - two_s * (i * i + k * k); X[5] = two_s * (j * k - i * r);
    X[6] = two_s * (i * k - j * r); X[7] = two_s * (j * k + i * r); X[8] = 1.f - two_s * (i * i + j * j);
    for (int d = 0; d < 3; ++d) {
      X[9 + d]  = o7[4 + d] * std_d[d] + mean_d[d];                  // translation
      X[12 + d] = centerFin[b * 16 + 12 + d] * std_p[d] + mean_p[d]; // center
    }
  }
}

// ---------------- K10: final outputs ----------------------------------
__global__ __launch_bounds__(256) void finalize_out(
    const _Float16* __restrict__ NR2, const _Float16* __restrict__ Eff,
    const float* __restrict__ w2, const float* __restrict__ b2,
    const float* __restrict__ attn_w, const float* __restrict__ attn_b,
    const float* __restrict__ xform, const float* __restrict__ state_cur,
    const float* __restrict__ p_inst, const float* __restrict__ touch,
    const float* __restrict__ mean_d, const float* __restrict__ std_d,
    float* __restrict__ out) {
  int i = blockIdx.x * blockDim.x + threadIdx.x;     // over BB*NP_
  if (i >= BB * NP_) return;
  int b = i / NP_, p = i - b * NP_;
  long row = (long)b * NN_ + p;
  float nr[3] = {b2[0], b2[1], b2[2]};
  float at = attn_b[0];
  const _Float16* e2 = NR2 + row * NF_;
  const _Float16* ef = Eff + row * NF_;
  for (int k = 0; k < 128; ++k) {
    float a = (float)e2[k];
    nr[0] += a * w2[k]; nr[1] += a * w2[128 + k]; nr[2] += a * w2[256 + k];
    at += (float)ef[k] * attn_w[k];
  }
  at = 1.f / (1.f + __expf(-at));
  const float* X = xform + b * 16;
  float p0[3], q[3];
  for (int d = 0; d < 3; ++d) { p0[d] = state_cur[row * 3 + d]; q[d] = p0[d] - X[12 + d]; }
  float pw = p_inst[(long)b * NP_ + p];
  float tch = touch[b];
  float* out_pos = out;
  float* out_mot = out + BB * NP_ * 3;
  float* out_att = out + 2 * BB * NP_ * 3;
  for (int d = 0; d < 3; ++d) {
    float p1 = q[0] * X[0 * 3 + d] + q[1] * X[1 * 3 + d] + q[2] * X[2 * 3 + d]
             + X[9 + d] + X[12 + d];
    float rigid = pw * ((p1 - p0[d]) - mean_d[d]) / std_d[d];
    float motion = at * nr[d] + rigid;
    float pm = tch * (motion * std_d[d] + mean_d[d]);
    out_mot[(long)i * 3 + d] = pm;
    float c = fminf(fmaxf(pm, -BOUND_), BOUND_);
    out_pos[(long)i * 3 + d] = p0[d] + c;
  }
  out_att[i] = at;
}

// ======================================================================
extern "C" void kernel_launch(void* const* d_in, const int* in_sizes, int n_in,
                              void* d_out, int out_size, void* d_ws, size_t ws_size,
                              hipStream_t stream) {
  // ---- inputs (setup_inputs dict order, param lists flattened) ----
  const float* attrs     = (const float*)d_in[0];
  const float* state_cur = (const float*)d_in[1];
  const float* action    = (const float*)d_in[2];
  const float* Rr        = (const float*)d_in[3];
  const float* Rs        = (const float*)d_in[4];
  const float* memory    = (const float*)d_in[5];
  const float* p_rigid   = (const float*)d_in[6];
  const float* p_inst    = (const float*)d_in[7];
  const float* physics   = (const float*)d_in[8];
  const float* mean_p    = (const float*)d_in[9];
  const float* std_p     = (const float*)d_in[10];
  const float* mean_d    = (const float*)d_in[11];
  const float* std_d     = (const float*)d_in[12];
  const float* pe_w0 = (const float*)d_in[13]; const float* pe_b0 = (const float*)d_in[14];
  const float* pe_w1 = (const float*)d_in[15]; const float* pe_b1 = (const float*)d_in[16];
  const float* pe_w2 = (const float*)d_in[17]; const float* pe_b2 = (const float*)d_in[18];
  const float* re_w0 = (const float*)d_in[19]; const float* re_b0 = (const float*)d_in[20];
  const float* re_w1 = (const float*)d_in[21]; const float* re_b1 = (const float*)d_in[22];
  const float* re_w2 = (const float*)d_in[23]; const float* re_b2 = (const float*)d_in[24];
  const float* pp_w  = (const float*)d_in[25]; const float* pp_b  = (const float*)d_in[26];
  const float* rp_w  = (const float*)d_in[27]; const float* rp_b  = (const float*)d_in[28];
  const float* rg_w0 = (const float*)d_in[29]; const float* rg_b0 = (const float*)d_in[30];
  const float* rg_w1 = (const float*)d_in[31]; const float* rg_b1 = (const float*)d_in[32];
  const float* rg_w2 = (const float*)d_in[33]; const float* rg_b2 = (const float*)d_in[34];
  const float* nr_w0 = (const float*)d_in[35]; const float* nr_b0 = (const float*)d_in[36];
  const float* nr_w1 = (const float*)d_in[37]; const float* nr_b1 = (const float*)d_in[38];
  const float* nr_w2 = (const float*)d_in[39]; const float* nr_b2 = (const float*)d_in[40];
  const float* attn_w = (const float*)d_in[41]; const float* attn_b = (const float*)d_in[42];
  (void)in_sizes; (void)n_in; (void)out_size; (void)ws_size;

  // ---- workspace bump allocator ----
  char* base = (char*)d_ws;
  size_t off = 0;
  auto alloc = [&](size_t bytes) -> void* {
    void* p = base + off;
    off = (off + bytes + 255) & ~(size_t)255;
    return p;
  };
  int*      ridx      = (int*)alloc((size_t)BR_ * 4);
  int*      sidx      = (int*)alloc((size_t)BR_ * 4);
  float*    touch     = (float*)alloc(BB * 4);
  float*    snorm     = (float*)alloc((size_t)BN_ * 15 * 4);
  float*    centerSum = (float*)alloc(BB * 16 * 4);
  float*    centerFin = (float*)alloc(BB * 16 * 4);
  float*    gvals     = (float*)alloc((size_t)BN_ * 4);
  _Float16* A16       = (_Float16*)alloc((size_t)BN_ * KPA_ * 2);
  _Float16* peW0f     = (_Float16*)alloc((size_t)NFH_ * KPA_ * 2);
  _Float16* peW1f     = (_Float16*)alloc((size_t)NFH_ * NFH_ * 2);
  _Float16* peW2f     = (_Float16*)alloc((size_t)NF_  * NFH_ * 2);
  _Float16* reW0f     = (_Float16*)alloc((size_t)NFH_ * KPR_ * 2);
  _Float16* reW1f     = (_Float16*)alloc((size_t)NFH_ * NFH_ * 2);
  _Float16* reW2f     = (_Float16*)alloc((size_t)NF_  * NFH_ * 2);
  _Float16* rpWf      = (_Float16*)alloc((size_t)NF_ * 384 * 2);
  _Float16* ppWf      = (_Float16*)alloc((size_t)NF_ * 256 * 2);
  _Float16* nrW0f     = (_Float16*)alloc((size_t)NF_ * NF_ * 2);
  _Float16* nrW1f     = (_Float16*)alloc((size_t)NF_ * NF_ * 2);
  _Float16* H1        = (_Float16*)alloc((size_t)BN_ * NFH_ * 2);
  _Float16* H2        = (_Float16*)alloc((size_t)BN_ * NFH_ * 2);
  _Float16* Penc      = (_Float16*)alloc((size_t)BN_ * NF_ * 2);
  _Float16* Eff       = (_Float16*)alloc((size_t)BN_ * NF_ * 2);
  _Float16* Rin       = (_Float16*)alloc((size_t)BR_ * KPR_ * 2);   // aliased by Erin
  _Float16* RH1       = (_Float16*)alloc((size_t)BR_ * NFH_ * 2);   // aliased by Erel
  _Float16* RH2       = (_Float16*)alloc((size_t)BR_ * NFH_ * 2);
  _Float16* Renc      = (_Float16*)alloc((size_t)BR_ * NF_ * 2);
  float*    Agg       = (float*)alloc((size_t)BN_ * NF_ * 4);
  _Float16* P2in      = (_Float16*)alloc((size_t)BN_ * 256 * 2);
  float*    instEff   = (float*)alloc(BB * NF_ * 4);
  float*    xform     = (float*)alloc(BB * 16 * 4);
  _Float16* Erin = Rin;       // [BR_,384] fits in Rin's [BR_,576] region
  _Float16* Erel = RH1;       // [BR_,128] fits in RH1's [BR_,256] region
  _Float16* NR1  = H1;        // nonrigid hiddens reuse p-encoder hiddens
  _Float16* NR2  = H2;

  auto gemm = [&](const _Float16* X, const _Float16* W, const float* bias,
                  _Float16* Y, int M, int Kp, int Nout, int relu) {
    dim3 g(M / 128, Nout / 32);     // wave = 32x32 tile, 4 waves/block
    gemm_f16_wmma<<<g, dim3(128), 0, stream>>>(X, W, bias, Y, M, Kp, Nout, relu);
  };
  auto cvt = [&](const float* src, _Float16* dst, int rows, int K, int Kp) {
    int n = rows * Kp;
    cvt_pad<<<(n + 255) / 256, 256, 0, stream>>>(src, dst, rows, K, Kp);
  };

  // ---- zero-init small accumulators ----
  hipMemsetAsync(touch, 0, BB * 4, stream);
  hipMemsetAsync(centerSum, 0, BB * 16 * 4, stream);

  // ---- decode one-hot relation matrices ----
  decode_onehot<<<BR_, 32, 0, stream>>>(Rr, ridx, nullptr);
  decode_onehot<<<BR_, 32, 0, stream>>>(Rs, sidx, touch);

  // ---- preprocessing ----
  build_state_norm<<<(BN_ + 255) / 256, 256, 0, stream>>>(
      state_cur, action, mean_p, std_p, mean_d, std_d, snorm);
  center_accum<<<BB, 256, 0, stream>>>(snorm, p_inst, centerSum);
  center_fin<<<1, 32, 0, stream>>>(centerSum, centerFin);
  build_A<<<BN_, 288, 0, stream>>>(attrs, physics, p_inst, p_rigid, snorm,
                                   centerFin, memory, A16, gvals);

  // ---- weight conversion (f32 -> f16, K padded) ----
  cvt(pe_w0, peW0f, NFH_, PIN_, KPA_);
  cvt(pe_w1, peW1f, NFH_, NFH_, NFH_);
  cvt(pe_w2, peW2f, NF_,  NFH_, NFH_);
  cvt(re_w0, reW0f, NFH_, RIN_, KPR_);
  cvt(re_w1, reW1f, NFH_, NFH_, NFH_);
  cvt(re_w2, reW2f, NF_,  NFH_, NFH_);
  cvt(rp_w,  rpWf,  NF_, 384, 384);
  cvt(pp_w,  ppWf,  NF_, 256, 256);
  cvt(nr_w0, nrW0f, NF_, NF_, NF_);
  cvt(nr_w1, nrW1f, NF_, NF_, NF_);

  // ---- particle encoder MLP (WMMA) ----
  gemm(A16, peW0f, pe_b0, H1, BN_, KPA_, NFH_, 1);
  gemm(H1,  peW1f, pe_b1, H2, BN_, NFH_, NFH_, 1);
  gemm(H2,  peW2f, pe_b2, Penc, BN_, NFH_, NF_, 1);

  // ---- relation encoder MLP (WMMA) ----
  build_rin<<<BR_, 192, 0, stream>>>(A16, snorm, gvals, ridx, sidx, Rin);
  gemm(Rin, reW0f, re_b0, RH1, BR_, KPR_, NFH_, 1);
  gemm(RH1, reW1f, re_b1, RH2, BR_, NFH_, NFH_, 1);
  gemm(RH2, reW2f, re_b2, Renc, BR_, NFH_, NF_, 1);

  // ---- message-passing loop ----
  hipMemcpyAsync(Eff, Penc, (size_t)BN_ * NF_ * 2, hipMemcpyDeviceToDevice, stream);
  for (int step = 0; step < 3; ++step) {
    build_erin<<<BR_, 128, 0, stream>>>(Renc, Eff, ridx, sidx, Erin);
    gemm(Erin, rpWf, rp_b, Erel, BR_, 384, NF_, 1);
    hipMemsetAsync(Agg, 0, (size_t)BN_ * NF_ * 4, stream);
    scatter_add<<<(BR_ * NF_ + 255) / 256, 256, 0, stream>>>(Erel, ridx, Agg);
    build_p2in<<<(BN_ * 256 + 255) / 256, 256, 0, stream>>>(Penc, Agg, P2in);
    gemm(P2in, ppWf, pp_b, Eff, BN_, 256, NF_, 1);
  }

  // ---- nonrigid hidden layers (WMMA) ----
  gemm(Eff, nrW0f, nr_b0, NR1, BN_, NF_, NF_, 1);
  gemm(NR1, nrW1f, nr_b1, NR2, BN_, NF_, NF_, 1);

  // ---- rigid branch ----
  inst_eff_reduce<<<BB, 128, 0, stream>>>(Eff, p_inst, instEff);
  rigid_mlp<<<BB, 128, 0, stream>>>(instEff, rg_w0, rg_b0, rg_w1, rg_b1,
                                    rg_w2, rg_b2, centerFin, mean_p, std_p,
                                    mean_d, std_d, xform);

  // ---- final outputs ----
  finalize_out<<<(BB * NP_ + 255) / 256, 256, 0, stream>>>(
      NR2, Eff, nr_w2, nr_b2, attn_w, attn_b, xform, state_cur, p_inst,
      touch, mean_d, std_d, (float*)d_out);
}
